// Encoder_59803124629550
// MI455X (gfx1250) — matrix operand order repaired and checked by reading, not compiled, not run on hardware
//
#include <hip/hip_runtime.h>
#include <hip/hip_bf16.h>
#include <stdint.h>

// ---------------- problem constants ----------------
constexpr int B  = 2;
constexpr int L  = 2048;
constexpr int H  = 12;
constexpr int NC = 512;
constexpr int D  = 768;
constexpr int KK = 50;            // top-k
constexpr int P  = KK * (KK - 1); // 2450 ordered pairs
constexpr int D3 = 3 * D;         // 2304
constexpr int NCOL = 57 + 20;     // 77 (relations + nota)
constexpr int M_CTX = KK * KK;    // 2500 rows of ctx GEMM
constexpr int MT_CTX  = (M_CTX + 15) / 16; // 157
constexpr int MT_PAIR = (P + 15) / 16;     // 154
constexpr int NPAIR_CTX = (D / 16) / 2;    // 24 pairs of N-tiles (48 tiles)
constexpr int NT_PAIR = (NCOL + 15) / 16;  // 5

typedef __attribute__((ext_vector_type(16))) __bf16 v16bf;
typedef __attribute__((ext_vector_type(8)))  __bf16 v8bf;
typedef __attribute__((ext_vector_type(8)))  float  v8f;

__device__ __forceinline__ unsigned short f2bf(float f) {
  unsigned u = __float_as_uint(f);
  // round-to-nearest-even bf16
  unsigned r = (u + 0x7FFFu + ((u >> 16) & 1u)) >> 16;
  return (unsigned short)r;
}

// ---------------- 1: span mean embeddings + anchor max score ----------------
__global__ __launch_bounds__(256) void k_span_embs(
    const float* __restrict__ seq, const int* __restrict__ starts,
    const int* __restrict__ lens, const float* __restrict__ anchor,
    float* __restrict__ embs, float* __restrict__ spanmax) {
  __shared__ float semb[D];
  __shared__ float sc[66];
  int bid = blockIdx.x;
  int b = bid / NC, n = bid % NC;
  int tid = threadIdx.x;
  int st = starts[b * NC + n];
  int ln = lens[b * NC + n];
  float w = 1.0f / (float)(ln + 1);
  const float* srow = seq + ((size_t)b * L + st) * D;
  for (int d = tid; d < D; d += 256) {
    float acc = 0.f;
    for (int j = 0; j <= ln; ++j) acc += srow[(size_t)j * D + d];
    float e = acc * w;
    embs[((size_t)b * NC + n) * D + d] = e;
    semb[d] = e;
  }
  __syncthreads();
  for (int c = tid; c < 66; c += 256) {
    const float* arow = anchor + (size_t)c * D;
    float dot = 0.f;
    for (int d = 0; d < D; ++d) dot += semb[d] * arow[d];
    sc[c] = dot;
  }
  __syncthreads();
  if (tid == 0) {
    float mx = sc[0];
    for (int c = 1; c < 66; ++c) mx = fmaxf(mx, sc[c]);
    spanmax[b * NC + n] = mx;
  }
}

// ---------------- 2: top-K (descending, stable ties -> lower index) --------
__global__ __launch_bounds__(512) void k_topk(const float* __restrict__ spanmax,
                                              int* __restrict__ topidx) {
  __shared__ float v[NC];
  __shared__ float rv[NC];
  __shared__ int   ri[NC];
  int b = blockIdx.x, tid = threadIdx.x;
  v[tid] = spanmax[b * NC + tid];
  __syncthreads();
  for (int k = 0; k < KK; ++k) {
    rv[tid] = v[tid]; ri[tid] = tid;
    __syncthreads();
    for (int s = NC / 2; s > 0; s >>= 1) {
      if (tid < s) {
        bool take = (rv[tid + s] > rv[tid]) ||
                    (rv[tid + s] == rv[tid] && ri[tid + s] < ri[tid]);
        if (take) { rv[tid] = rv[tid + s]; ri[tid] = ri[tid + s]; }
      }
      __syncthreads();
    }
    if (tid == 0) {
      topidx[b * KK + k] = ri[0];
      v[ri[0]] = -__builtin_inff();
    }
    __syncthreads();
  }
}

// ---------------- 3: gather selected embeddings -> bf16 --------------------
__global__ __launch_bounds__(256) void k_gather_ent(
    const float* __restrict__ embs, const int* __restrict__ topidx,
    unsigned short* __restrict__ entbf) {
  int bid = blockIdx.x;
  int b = bid / KK, k = bid % KK;
  int n = topidx[b * KK + k];
  const float* src = embs + ((size_t)b * NC + n) * D;
  unsigned short* dst = entbf + ((size_t)b * KK + k) * D;
  for (int d = threadIdx.x; d < D; d += 256) dst[d] = f2bf(src[d]);
}

// ---------------- 4: seq -> bf16 transposed [B][D][L] ----------------------
__global__ __launch_bounds__(256) void k_seqT(const float* __restrict__ seq,
                                              unsigned short* __restrict__ seqT) {
  size_t i = (size_t)blockIdx.x * 256 + threadIdx.x;
  if (i >= (size_t)B * L * D) return;
  int b = (int)(i / ((size_t)L * D));
  size_t rem = i - (size_t)b * L * D;
  int l = (int)(rem / D), d = (int)(rem % D);
  seqT[((size_t)b * D + d) * L + l] = f2bf(seq[i]);
}

// ---------------- 5: [relation|nota] -> bf16 [77][2304] --------------------
__global__ __launch_bounds__(256) void k_rn(const float* __restrict__ rel,
                                            const float* __restrict__ nota,
                                            unsigned short* __restrict__ rn) {
  int i = blockIdx.x * 256 + threadIdx.x;
  if (i >= NCOL * D3) return;
  int n = i / D3, kk = i % D3;
  float v = (n < 57) ? rel[(size_t)n * D3 + kk] : nota[(size_t)(n - 57) * D3 + kk];
  rn[i] = f2bf(v);
}

// ---------------- 6: att_k[b,k,h,m] (sparse span rows) ---------------------
__global__ __launch_bounds__(256) void k_attk(
    const float* __restrict__ att, const int* __restrict__ topidx,
    const int* __restrict__ starts, const int* __restrict__ lens,
    float* __restrict__ attk) {
  int bid = blockIdx.x;
  int b = bid / (KK * H);
  int r = bid % (KK * H);
  int k = r / H, h = r % H;
  int n = topidx[b * KK + k];
  int st = starts[b * NC + n];
  int ln = lens[b * NC + n];
  float w = 1.0f / (float)(ln + 1);
  const float* arow = att + (((size_t)b * H + h) * L + st) * L;
  float* dst = attk + (((size_t)b * KK + k) * H + h) * L;
  for (int m = threadIdx.x; m < L; m += 256) {
    float acc = 0.f;
    for (int j = 0; j <= ln; ++j) acc += arow[(size_t)j * L + m];
    dst[m] = acc * w;
  }
}

// ---------------- 7: A = sum_h attk_s*attk_o; normalize; -> bf16 -----------
__global__ __launch_bounds__(256) void k_A(const float* __restrict__ attk,
                                           unsigned short* __restrict__ abf) {
  __shared__ float red[256];
  int bid = blockIdx.x;
  int b = bid / (KK * KK);
  int r = bid % (KK * KK);
  int s = r / KK, o = r % KK;
  int tid = threadIdx.x;
  const float* as = attk + ((size_t)b * KK + s) * H * L;
  const float* ao = attk + ((size_t)b * KK + o) * H * L;
  float av[L / 256];
  float part = 0.f;
  #pragma unroll
  for (int c = 0; c < L / 256; ++c) {
    int m = c * 256 + tid;
    float acc = 0.f;
    #pragma unroll
    for (int h = 0; h < H; ++h) acc += as[(size_t)h * L + m] * ao[(size_t)h * L + m];
    av[c] = acc;
    part += acc;
  }
  red[tid] = part;
  __syncthreads();
  for (int st2 = 128; st2 > 0; st2 >>= 1) {
    if (tid < st2) red[tid] += red[tid + st2];
    __syncthreads();
  }
  float inv = 1.0f / red[0];
  unsigned short* dst = abf + ((size_t)b * M_CTX + (size_t)s * KK + o) * L;
  #pragma unroll
  for (int c = 0; c < L / 256; ++c) dst[c * 256 + tid] = f2bf(av[c] * inv);
}

// ---------------- 8: ctx GEMM via WMMA bf16 (f32 accum) --------------------
// C[b, m=s*50+o, n] = sum_k a_bf[b,m,k] * seqT[b,n,k]   (m<2500, n<768, K=2048)
__global__ __launch_bounds__(32) void k_ctx_wmma(
    const unsigned short* __restrict__ abf_, const unsigned short* __restrict__ seqT_,
    unsigned short* __restrict__ ctxbf) {
  const __bf16* abf  = (const __bf16*)abf_;
  const __bf16* seqT = (const __bf16*)seqT_;
  int bid = blockIdx.x;
  int b  = bid / (MT_CTX * NPAIR_CTX);
  int r  = bid % (MT_CTX * NPAIR_CTX);
  int mT = r / NPAIR_CTX, nP = r % NPAIR_CTX;
  int lane = threadIdx.x;
  int half = lane >> 4, row = lane & 15;
  int m = mT * 16 + row; if (m > M_CTX - 1) m = M_CTX - 1;
  const __bf16* Arow = abf + ((size_t)b * M_CTX + m) * L;
  int n0 = (nP * 2 + 0) * 16 + row;
  int n1 = (nP * 2 + 1) * 16 + row;
  const __bf16* Bp0 = seqT + ((size_t)b * D + n0) * L + 8 * half;
  const __bf16* Bp1 = seqT + ((size_t)b * D + n1) * L + 8 * half;
  v8f c0 = {0.f,0.f,0.f,0.f,0.f,0.f,0.f,0.f};
  v8f c1 = c0;
  for (int k0 = 0; k0 < L; k0 += 32) {
    __builtin_prefetch(Arow + k0 + 128, 0, 1);
    v8bf a0 = *(const v8bf*)(Arow + k0 + 8 * half);
    v8bf a1 = *(const v8bf*)(Arow + k0 + 16 + 8 * half);
    v16bf af;
    #pragma unroll
    for (int i = 0; i < 8; ++i) { af[i] = a0[i]; af[i + 8] = a1[i]; }
    v8bf b00 = *(const v8bf*)(Bp0 + k0);
    v8bf b01 = *(const v8bf*)(Bp0 + k0 + 16);
    v8bf b10 = *(const v8bf*)(Bp1 + k0);
    v8bf b11 = *(const v8bf*)(Bp1 + k0 + 16);
    v16bf bf0, bf1;
    #pragma unroll
    for (int i = 0; i < 8; ++i) { bf0[i] = b00[i]; bf0[i + 8] = b01[i]; bf1[i] = b10[i]; bf1[i + 8] = b11[i]; }
    c0 = __builtin_amdgcn_wmma_f32_16x16x32_bf16(false, af, false, bf0, (short)0, c0, false, false);
    c1 = __builtin_amdgcn_wmma_f32_16x16x32_bf16(false, af, false, bf1, (short)0, c1, false, false);
  }
  #pragma unroll
  for (int i = 0; i < 8; ++i) {
    int mm = mT * 16 + i + 8 * half;
    if (mm < M_CTX) {
      size_t base = ((size_t)b * M_CTX + mm) * D;
      ctxbf[base + n0] = f2bf(c0[i]);
      ctxbf[base + n1] = f2bf(c1[i]);
    }
  }
}

// ---------------- 9: pair scoring GEMM via WMMA bf16 -----------------------
// raw[b,p,n] = sum_kk concat[b,p,kk] * RN[n,kk]; concat = [eS | eO | ctx(s,o)]
__global__ __launch_bounds__(32) void k_pair_wmma(
    const unsigned short* __restrict__ entbf_, const unsigned short* __restrict__ ctxbf_,
    const unsigned short* __restrict__ rn_, float* __restrict__ raw) {
  const __bf16* entbf = (const __bf16*)entbf_;
  const __bf16* ctxbf = (const __bf16*)ctxbf_;
  const __bf16* rn    = (const __bf16*)rn_;
  int bid = blockIdx.x;
  int b  = bid / (MT_PAIR * NT_PAIR);
  int r  = bid % (MT_PAIR * NT_PAIR);
  int mT = r / NT_PAIR, nT = r % NT_PAIR;
  int lane = threadIdx.x;
  int half = lane >> 4, row = lane & 15;
  int p = mT * 16 + row; if (p > P - 1) p = P - 1;
  int s  = p / (KK - 1);
  int rr = p % (KK - 1);
  int o  = rr + (rr >= s ? 1 : 0);
  const __bf16* eS = entbf + ((size_t)b * KK + s) * D;
  const __bf16* eO = entbf + ((size_t)b * KK + o) * D;
  const __bf16* eC = ctxbf + ((size_t)b * M_CTX + (size_t)s * KK + o) * D;
  int ncol = nT * 16 + row;
  int ncl = ncol > NCOL - 1 ? NCOL - 1 : ncol;
  const __bf16* Bp = rn + (size_t)ncl * D3 + 8 * half;
  v8f c = {0.f,0.f,0.f,0.f,0.f,0.f,0.f,0.f};
  for (int k0 = 0; k0 < D3; k0 += 32) {
    int ka = k0 + 8 * half;        // 8-element run 0
    int kb = ka + 16;              // 8-element run 1
    const __bf16* pa = (ka < D) ? (eS + ka) : (ka < 2 * D) ? (eO + ka - D) : (eC + ka - 2 * D);
    const __bf16* pb = (kb < D) ? (eS + kb) : (kb < 2 * D) ? (eO + kb - D) : (eC + kb - 2 * D);
    v8bf a0 = *(const v8bf*)pa;
    v8bf a1 = *(const v8bf*)pb;
    v16bf af;
    #pragma unroll
    for (int i = 0; i < 8; ++i) { af[i] = a0[i]; af[i + 8] = a1[i]; }
    v8bf b0 = *(const v8bf*)(Bp + k0);
    v8bf b1 = *(const v8bf*)(Bp + k0 + 16);
    v16bf bf;
    #pragma unroll
    for (int i = 0; i < 8; ++i) { bf[i] = b0[i]; bf[i + 8] = b1[i]; }
    c = __builtin_amdgcn_wmma_f32_16x16x32_bf16(false, af, false, bf, (short)0, c, false, false);
  }
  #pragma unroll
  for (int i = 0; i < 8; ++i) {
    int pp = mT * 16 + i + 8 * half;
    if (pp < P && ncol < NCOL)
      raw[((size_t)b * P + pp) * NCOL + ncol] = c[i];
  }
}

// ---------------- 10: assemble output [B,P,58] -----------------------------
__global__ __launch_bounds__(64) void k_out(const float* __restrict__ raw,
                                            float* __restrict__ out) {
  int bid = blockIdx.x;
  int b = bid / P, p = bid % P;
  const float* src = raw + ((size_t)b * P + p) * NCOL;
  float* dst = out + ((size_t)b * P + p) * 58;
  int t = threadIdx.x;
  if (t < 57) dst[1 + t] = src[t];
  if (t == 0) {
    float mx = src[57];
    for (int j = 58; j < NCOL; ++j) mx = fmaxf(mx, src[j]);
    dst[0] = mx;
  }
}

// ---------------- host launcher --------------------------------------------
extern "C" void kernel_launch(void* const* d_in, const int* in_sizes, int n_in,
                              void* d_out, int out_size, void* d_ws, size_t ws_size,
                              hipStream_t stream) {
  (void)in_sizes; (void)n_in; (void)out_size; (void)ws_size;
  const float* seq    = (const float*)d_in[0];
  const float* att    = (const float*)d_in[1];
  const int*   starts = (const int*)d_in[2];
  const int*   lens   = (const int*)d_in[3];
  const float* anchor = (const float*)d_in[4];
  const float* rel    = (const float*)d_in[5];
  const float* nota   = (const float*)d_in[6];
  float* out = (float*)d_out;

  uint8_t* w = (uint8_t*)d_ws;
  auto alloc = [&](size_t bytes) -> void* {
    void* pp = (void*)w;
    w += (bytes + 255) & ~(size_t)255;
    return pp;
  };
  float*          embs    = (float*)alloc((size_t)B * NC * D * 4);
  float*          spanmax = (float*)alloc((size_t)B * NC * 4);
  int*            topidx  = (int*)alloc((size_t)B * KK * 4);
  unsigned short* entbf   = (unsigned short*)alloc((size_t)B * KK * D * 2);
  unsigned short* seqT    = (unsigned short*)alloc((size_t)B * D * L * 2);
  unsigned short* rnbf    = (unsigned short*)alloc((size_t)NCOL * D3 * 2);
  float*          attk    = (float*)alloc((size_t)B * KK * H * L * 4);
  unsigned short* abf     = (unsigned short*)alloc((size_t)B * M_CTX * L * 2);
  unsigned short* ctxbf   = (unsigned short*)alloc((size_t)B * M_CTX * D * 2);
  float*          raw     = (float*)alloc((size_t)B * P * NCOL * 4);

  k_span_embs<<<B * NC, 256, 0, stream>>>(seq, starts, lens, anchor, embs, spanmax);
  k_topk<<<B, NC, 0, stream>>>(spanmax, topidx);
  k_gather_ent<<<B * KK, 256, 0, stream>>>(embs, topidx, entbf);
  k_seqT<<<(int)(((size_t)B * L * D + 255) / 256), 256, 0, stream>>>(seq, seqT);
  k_rn<<<(NCOL * D3 + 255) / 256, 256, 0, stream>>>(rel, nota, rnbf);
  k_attk<<<B * KK * H, 256, 0, stream>>>(att, topidx, starts, lens, attk);
  k_A<<<B * KK * KK, 256, 0, stream>>>(attk, abf);
  k_ctx_wmma<<<B * MT_CTX * NPAIR_CTX, 32, 0, stream>>>(abf, seqT, ctxbf);
  k_pair_wmma<<<B * MT_PAIR * NT_PAIR, 32, 0, stream>>>(entbf, ctxbf, rnbf, raw);
  k_out<<<B * P, 64, 0, stream>>>(raw, out);
}
